// AddingProblemIndRNNModel_1614907703319
// MI455X (gfx1250) — matrix operand order repaired and checked by reading, not compiled, not run on hardware
//
#include <hip/hip_runtime.h>

typedef _Float16 v8h  __attribute__((ext_vector_type(8)));
typedef _Float16 v16h __attribute__((ext_vector_type(16)));
typedef float    v8f  __attribute__((ext_vector_type(8)));

#define T_STEPS 4096
#define BATCH   128
#define HID     256

// ---------- kernel 0: convert W2 (fp32 -> fp16), row-major [n][k] kept ----------
__global__ void k_cvt_w2(const float* __restrict__ w2, _Float16* __restrict__ w2h) {
    int i = blockIdx.x * blockDim.x + threadIdx.x;
    w2h[i] = (_Float16)w2[i];
}

// ---------- kernel 1: input projection + layer-1 elementwise IndRNN scan ----------
// One thread per (b,h) channel; H1 stored time-major [t][b][h] in fp16 (WMMA A operand).
__global__ void k_layer1(const float* __restrict__ x, const float* __restrict__ W1,
                         const float* __restrict__ b1, const float* __restrict__ u1,
                         _Float16* __restrict__ H1) {
    int tid = blockIdx.x * blockDim.x + threadIdx.x;   // 0 .. B*H-1
    int h = tid & (HID - 1);
    int b = tid >> 8;
    float w0 = W1[2*h], w1 = W1[2*h + 1];
    float bb = b1[h],   uu = u1[h];
    const float2* xb = (const float2*)(x + (size_t)b * T_STEPS * 2);
    _Float16* dst = H1 + (size_t)b * HID + h;          // stride B*H per timestep
    float hs = 0.f;
    for (int t = 0; t < T_STEPS; ++t) {
        float2 xv = xb[t];
        float pre = fmaf(xv.x, w0, fmaf(xv.y, w1, fmaf(uu, hs, bb)));
        hs = pre > 0.f ? pre : 0.f;
        dst[(size_t)t * (BATCH * HID)] = (_Float16)hs;
    }
}

// ---------- kernel 2: Z = H1 @ W2^T  (M = T*B = 524288, N = K = 256), fp16 out ----------
// One wave per 2x2 group of 16x16 C tiles: A/B fragments each reused twice,
// 4 x v_wmma_f32_16x16x32_f16 per K-step, K-loop of 8 steps fully unrolled.
__global__ __launch_bounds__(256) void k_gemm(const _Float16* __restrict__ H1,
                                              const _Float16* __restrict__ W2h,
                                              _Float16* __restrict__ Zh) {
    const int lane  = threadIdx.x & 31;
    const int wave  = threadIdx.x >> 5;
    const unsigned wid = blockIdx.x * 8u + (unsigned)wave;
    const unsigned nP  = wid & 7u;    // 8 column-pair groups: cols [nP*32, nP*32+31]
    const unsigned mP  = wid >> 3;    // 16384 row-pair groups: rows [mP*32, mP*32+31]
    const unsigned r16 = lane & 15;
    const unsigned hi  = lane >> 4;   // 0 or 1

    // A fragments (16x32 f16): lane holds row r16, K chunks at hi*8 and hi*8+16
    const _Float16* A0 = H1  + ((size_t)mP * 32 + r16) * HID + hi * 8;
    const _Float16* A1 = A0 + (size_t)16 * HID;
    // B fragments (32x16 f16): lane holds col r16, K chunks at hi*8 and hi*8+16
    const _Float16* B0 = W2h + ((size_t)nP * 32 + r16) * HID + hi * 8;
    const _Float16* B1 = B0 + (size_t)16 * HID;

    v8f acc00 = {}, acc01 = {}, acc10 = {}, acc11 = {};
#pragma unroll
    for (int k = 0; k < HID; k += 32) {
        v16h av0 = __builtin_shufflevector(*(const v8h*)(A0 + k), *(const v8h*)(A0 + k + 16),
                                           0,1,2,3,4,5,6,7,8,9,10,11,12,13,14,15);
        v16h av1 = __builtin_shufflevector(*(const v8h*)(A1 + k), *(const v8h*)(A1 + k + 16),
                                           0,1,2,3,4,5,6,7,8,9,10,11,12,13,14,15);
        v16h bv0 = __builtin_shufflevector(*(const v8h*)(B0 + k), *(const v8h*)(B0 + k + 16),
                                           0,1,2,3,4,5,6,7,8,9,10,11,12,13,14,15);
        v16h bv1 = __builtin_shufflevector(*(const v8h*)(B1 + k), *(const v8h*)(B1 + k + 16),
                                           0,1,2,3,4,5,6,7,8,9,10,11,12,13,14,15);
        acc00 = __builtin_amdgcn_wmma_f32_16x16x32_f16(false, av0, false, bv0, (short)0, acc00, false, false);
        acc01 = __builtin_amdgcn_wmma_f32_16x16x32_f16(false, av0, false, bv1, (short)0, acc01, false, false);
        acc10 = __builtin_amdgcn_wmma_f32_16x16x32_f16(false, av1, false, bv0, (short)0, acc10, false, false);
        acc11 = __builtin_amdgcn_wmma_f32_16x16x32_f16(false, av1, false, bv1, (short)0, acc11, false, false);
    }

    // C layout per tile: acc[v] -> row (mBase + hi*8 + v), col (nBase + r16)
    const unsigned mB = mP * 32, nB = nP * 32;
    {
        _Float16* p = Zh + ((size_t)mB + hi * 8) * HID + nB + r16;
#pragma unroll
        for (int v = 0; v < 8; ++v) p[(size_t)v * HID] = (_Float16)acc00[v];
    }
    {
        _Float16* p = Zh + ((size_t)mB + hi * 8) * HID + nB + 16 + r16;
#pragma unroll
        for (int v = 0; v < 8; ++v) p[(size_t)v * HID] = (_Float16)acc01[v];
    }
    {
        _Float16* p = Zh + ((size_t)mB + 16 + hi * 8) * HID + nB + r16;
#pragma unroll
        for (int v = 0; v < 8; ++v) p[(size_t)v * HID] = (_Float16)acc10[v];
    }
    {
        _Float16* p = Zh + ((size_t)mB + 16 + hi * 8) * HID + nB + 16 + r16;
#pragma unroll
        for (int v = 0; v < 8; ++v) p[(size_t)v * HID] = (_Float16)acc11[v];
    }
}

// ---------- kernel 3: layer-2 elementwise scan + final readout ----------
// Block b handles batch element b; thread n handles channel n. Coalesced fp16 reads.
// hs carried at fp32 in registers -> no error accumulation from fp16 z storage.
__global__ void k_layer2_scan(const _Float16* __restrict__ Zh,
                              const float* __restrict__ b2, const float* __restrict__ u2,
                              const float* __restrict__ Wf, const float* __restrict__ bf,
                              float* __restrict__ out) {
    const int n = threadIdx.x;     // 0..255
    const int b = blockIdx.x;      // 0..127
    const float bb = b2[n], uu = u2[n];
    const _Float16* zp = Zh + (size_t)b * HID + n;
    float hs = 0.f;
#pragma unroll 8
    for (int t = 0; t < T_STEPS; ++t) {
        float z = (float)zp[(size_t)t * (BATCH * HID)];
        float pre = z + fmaf(uu, hs, bb);
        hs = pre > 0.f ? pre : 0.f;
    }
    __shared__ float red[HID];
    red[n] = hs * Wf[n];
    __syncthreads();
    for (int s = HID / 2; s > 0; s >>= 1) {
        if (n < s) red[n] += red[n + s];
        __syncthreads();
    }
    if (n == 0) out[b] = red[0] + bf[0];
}

extern "C" void kernel_launch(void* const* d_in, const int* in_sizes, int n_in,
                              void* d_out, int out_size, void* d_ws, size_t ws_size,
                              hipStream_t stream) {
    const float* x  = (const float*)d_in[0];
    const float* W1 = (const float*)d_in[1];
    const float* b1 = (const float*)d_in[2];
    const float* u1 = (const float*)d_in[3];
    const float* W2 = (const float*)d_in[4];
    const float* b2 = (const float*)d_in[5];
    const float* u2 = (const float*)d_in[6];
    const float* Wf = (const float*)d_in[7];
    const float* bf = (const float*)d_in[8];
    float* out = (float*)d_out;

    // Workspace layout: Zh (fp16, 256 MB) | H1 (fp16, 256 MB) | W2h (fp16, 128 KB)
    char* ws = (char*)d_ws;
    const size_t elems = (size_t)T_STEPS * BATCH * HID;   // 134,217,728
    _Float16* Zh  = (_Float16*)ws;
    _Float16* H1  = (_Float16*)(ws + elems * sizeof(_Float16));
    _Float16* W2h = (_Float16*)(ws + 2 * elems * sizeof(_Float16));

    k_cvt_w2<<<(HID * HID) / 256, 256, 0, stream>>>(W2, W2h);
    k_layer1<<<(BATCH * HID) / 256, 256, 0, stream>>>(x, W1, b1, u1, H1);

    // waves = (M/32) * (N/32) = 16384 * 8 = 131072; 8 waves/block
    k_gemm<<<131072 / 8, 256, 0, stream>>>(H1, W2h, Zh);

    k_layer2_scan<<<BATCH, HID, 0, stream>>>(Zh, b2, u2, Wf, bf, out);
}
